// HOGRL_46377056862933
// MI455X (gfx1250) — compile-verified, matrix operand order checked by reading, not run on hardware
//
#include <hip/hip_runtime.h>
#include <hip/hip_bf16.h>

typedef __attribute__((ext_vector_type(16))) __bf16 v16bf;
typedef __attribute__((ext_vector_type(8)))  __bf16 v8bf;
typedef __attribute__((ext_vector_type(8)))  float  v8f;

#define D_IN   256
#define H_DIM  128
#define C4     512   // 4 fused channels * 128

__device__ __forceinline__ __bf16 f2bf(float f) {
    unsigned u = __builtin_bit_cast(unsigned, f);
    unsigned r = u + 0x7FFFu + ((u >> 16) & 1u);        // round-to-nearest-even
    unsigned short h = (unsigned short)(r >> 16);
    return __builtin_bit_cast(__bf16, h);
}

// Load one WMMA 16-bit operand slice for this lane: elements 0..7 = K-chunk at p,
// elements 8..15 = K-chunk at p+16 (per CDNA5 16-bit A-matrix 16x32 layout).
__device__ __forceinline__ v16bf ld16(const __bf16* p) {
    v8bf lo = *(const v8bf*)p;
    v8bf hi = *(const v8bf*)(p + 16);
    v16bf r;
#pragma unroll
    for (int i = 0; i < 8; ++i) { r[i] = lo[i]; r[i + 8] = hi[i]; }
    return r;
}

// ---------------- prep kernels ----------------

__global__ void k_zero_f4(float4* __restrict__ p, long n4) {
    long i = (long)blockIdx.x * blockDim.x + threadIdx.x;
    long stride = (long)gridDim.x * blockDim.x;
    float4 z = {0.f, 0.f, 0.f, 0.f};
    for (; i < n4; i += stride) p[i] = z;
}

__global__ void k_cvt_x(const float* __restrict__ x, __bf16* __restrict__ xb, long n) {
    long i = (long)blockIdx.x * blockDim.x + threadIdx.x;
    long stride = (long)gridDim.x * blockDim.x;
    for (; i < n; i += stride) xb[i] = f2bf(x[i]);
}

// WcatT[j][k] = column j of [W_ord0|W_ord1|W_ord2|W_orig] at input-dim k (bf16, col-major pack)
__global__ void k_pack_wcat(const float* __restrict__ Word, const float* __restrict__ bord,
                            const float* __restrict__ Worig, const float* __restrict__ borig,
                            __bf16* __restrict__ WcatT, float* __restrict__ bcat) {
    int tid = blockIdx.x * blockDim.x + threadIdx.x;   // 512*256 threads
    if (tid >= C4 * D_IN) return;
    int j = tid >> 8;          // output column 0..511
    int k = tid & 255;         // input dim 0..255
    int ord = j >> 7;          // 0..3
    int h = j & 127;
    float v = (ord < 3) ? Word[(size_t)ord * D_IN * H_DIM + (size_t)k * H_DIM + h]
                        : Worig[(size_t)k * H_DIM + h];
    WcatT[(size_t)j * D_IN + k] = f2bf(v);
    if (k == 0) bcat[j] = (ord < 3) ? bord[ord * H_DIM + h] : borig[h];
}

// W1T[n][k] = W1[k][n]  (bf16 col-major pack, 128x256)
__global__ void k_pack_w1(const float* __restrict__ W1, __bf16* __restrict__ W1T) {
    int tid = blockIdx.x * blockDim.x + threadIdx.x;   // 128*256 threads
    if (tid >= H_DIM * 2 * H_DIM) return;
    int n = tid >> 8, k = tid & 255;
    W1T[(size_t)n * 256 + k] = f2bf(W1[(size_t)k * H_DIM + n]);
}

// ---------------- GEMM 1: Hall[N,512] (bf16) = xb @ Wcat + bcat ----------------
// block = 256 threads (8 waves). Block -> one 16-row tile; wave w -> 64 cols.

__global__ void __launch_bounds__(256) k_gemm1(const __bf16* __restrict__ xb,
                                               const __bf16* __restrict__ WcatT,
                                               const float* __restrict__ bcat,
                                               __bf16* __restrict__ Hall, int N) {
    int wave = threadIdx.x >> 5;
    int lane = threadIdx.x & 31;
    int row0 = blockIdx.x * 16;
    if (row0 >= N) return;
    int col0 = wave * 64;
    int m = lane & 15, hi = lane >> 4;

    v8f acc[4] = {};
    const __bf16* arow = xb + (size_t)(row0 + m) * D_IN + hi * 8;
#pragma unroll
    for (int kt = 0; kt < 8; ++kt) {
        int ko = kt * 32;
        v16bf a = ld16(arow + ko);
#pragma unroll
        for (int sub = 0; sub < 4; ++sub) {
            const __bf16* bp = WcatT + (size_t)(col0 + sub * 16 + m) * D_IN + ko + hi * 8;
            v16bf b = ld16(bp);
            acc[sub] = __builtin_amdgcn_wmma_f32_16x16x32_bf16(
                false, a, false, b, (short)0, acc[sub], false, false);
        }
    }
#pragma unroll
    for (int sub = 0; sub < 4; ++sub) {
        int n = col0 + sub * 16 + m;
        float bias = bcat[n];
#pragma unroll
        for (int g = 0; g < 8; ++g) {
            int row = row0 + g + 8 * hi;
            Hall[(size_t)row * C4 + n] = f2bf(acc[sub][g] + bias);
        }
    }
}

// ---------------- SpMM: Hsum[row] += Hall[col] over edges (4 channels fused) ----
// One wave per edge: 2x uint4 (8 bf16) coalesced gathers per lane, exact bf16->f32
// expansion via shifts, fp32 accumulation via global_atomic_add_f32.

__global__ void k_spmm(const int* __restrict__ eidx, const __bf16* __restrict__ Hall,
                       float* __restrict__ Hsum, int E) {
    long gid = (long)blockIdx.x * blockDim.x + threadIdx.x;
    int e = (int)(gid >> 5);
    int lane = (int)(gid & 31);
    if (e >= E) return;
    int row = eidx[e];           // edge_index[0][e]
    int col = eidx[(size_t)E + e];
    const uint4* src = (const uint4*)(Hall + (size_t)col * C4);   // 64 uint4 per row
    float* dst = Hsum + (size_t)row * C4;
#pragma unroll
    for (int k = 0; k < 2; ++k) {
        uint4 u = src[k * 32 + lane];
        int base = (k * 32 + lane) * 8;
        unsigned w[4] = {u.x, u.y, u.z, u.w};
#pragma unroll
        for (int j = 0; j < 4; ++j) {
            float lo = __builtin_bit_cast(float, w[j] << 16);            // bf16 -> f32 exact
            float hi = __builtin_bit_cast(float, w[j] & 0xFFFF0000u);
            __hip_atomic_fetch_add(dst + base + 2 * j,     lo, __ATOMIC_RELAXED, __HIP_MEMORY_SCOPE_AGENT);
            __hip_atomic_fetch_add(dst + base + 2 * j + 1, hi, __ATOMIC_RELAXED, __HIP_MEMORY_SCOPE_AGENT);
        }
    }
}

// ---------------- Epilogue: relu, gate softmax, mixture, build z (bf16) ----------
// One wave per node.

__global__ void k_epilogue(const float* __restrict__ Hsum, const float* __restrict__ Wgate,
                           const float* __restrict__ bgate, const float* __restrict__ gammap,
                           __bf16* __restrict__ zb, int N) {
    long gid = (long)blockIdx.x * blockDim.x + threadIdx.x;
    int node = (int)(gid >> 5);
    int lane = (int)(gid & 31);
    if (node >= N) return;
    const float* hs = Hsum + (size_t)node * C4;

    float ex[3][4], ho[4], s[3];
#pragma unroll
    for (int ord = 0; ord < 3; ++ord) {
        float p = 0.f;
#pragma unroll
        for (int k = 0; k < 4; ++k) {
            int c = lane + 32 * k;
            float v = fmaxf(hs[ord * H_DIM + c], 0.f);
            ex[ord][k] = v;
            p += v * Wgate[ord * H_DIM + c];
        }
#pragma unroll
        for (int off = 16; off > 0; off >>= 1) p += __shfl_xor(p, off, 32);
        s[ord] = p + bgate[ord];
    }
#pragma unroll
    for (int k = 0; k < 4; ++k) ho[k] = fmaxf(hs[3 * H_DIM + lane + 32 * k], 0.f);

    float mx = fmaxf(fmaxf(s[0], s[1]), s[2]);
    float e0 = __expf(s[0] - mx), e1 = __expf(s[1] - mx), e2 = __expf(s[2] - mx);
    float inv = 1.f / (e0 + e1 + e2);
    float w0 = e0 * inv, w1 = e1 * inv, w2 = e2 * inv;
    float g = gammap[0];

    __bf16* zrow = zb + (size_t)node * (2 * H_DIM);
#pragma unroll
    for (int k = 0; k < 4; ++k) {
        int c = lane + 32 * k;
        zrow[c] = f2bf(ho[k]);
        float hh = w0 * ex[0][k] + w1 * ex[1][k] + w2 * ex[2][k];
        zrow[H_DIM + c] = f2bf(g * hh);
    }
}

// ---------------- GEMM 2 + classifier: out = relu(z@W1+b1) @ W2 + b2 ----------
// block = 128 threads (4 waves); each wave: 16 rows x 128 cols via 8 WMMA accumulators,
// then reduce h (16x128 in LDS) against W2 (128x2).

__global__ void __launch_bounds__(128) k_gemm2(const __bf16* __restrict__ zb,
                                               const __bf16* __restrict__ W1T,
                                               const float* __restrict__ b1,
                                               const float* __restrict__ W2,
                                               const float* __restrict__ b2,
                                               float* __restrict__ out, int N) {
    __shared__ float lds[4][16][132];   // stride 132 -> conflict-free column sweeps
    int wave = threadIdx.x >> 5;
    int lane = threadIdx.x & 31;
    int rt = blockIdx.x * 4 + wave;
    int row0 = rt * 16;
    bool active = (row0 < N);
    int m = lane & 15, hi = lane >> 4;

    if (active) {
        v8f acc[8] = {};
        const __bf16* arow = zb + (size_t)(row0 + m) * 256 + hi * 8;
#pragma unroll
        for (int kt = 0; kt < 8; ++kt) {
            int ko = kt * 32;
            v16bf a = ld16(arow + ko);
#pragma unroll
            for (int sub = 0; sub < 8; ++sub) {
                const __bf16* bp = W1T + (size_t)(sub * 16 + m) * 256 + ko + hi * 8;
                v16bf b = ld16(bp);
                acc[sub] = __builtin_amdgcn_wmma_f32_16x16x32_bf16(
                    false, a, false, b, (short)0, acc[sub], false, false);
            }
        }
#pragma unroll
        for (int sub = 0; sub < 8; ++sub) {
            int n = sub * 16 + m;
            float bias = b1[n];
#pragma unroll
            for (int g = 0; g < 8; ++g)
                lds[wave][g + 8 * hi][n] = fmaxf(acc[sub][g] + bias, 0.f);
        }
    }
    __syncthreads();
    if (active) {
        int o = lane >> 4;      // output class 0/1
        int r = lane & 15;      // row in tile
        float sum = 0.f;
#pragma unroll 8
        for (int c = 0; c < 128; ++c) sum += lds[wave][r][c] * W2[c * 2 + o];
        out[(size_t)(row0 + r) * 2 + o] = sum + b2[o];
    }
}

// ---------------- host launcher ----------------

extern "C" void kernel_launch(void* const* d_in, const int* in_sizes, int n_in,
                              void* d_out, int out_size, void* d_ws, size_t ws_size,
                              hipStream_t stream) {
    const float* x      = (const float*)d_in[0];
    const int*   eidx   = (const int*)  d_in[1];
    const float* Word   = (const float*)d_in[2];
    const float* bord   = (const float*)d_in[3];
    const float* Wgate  = (const float*)d_in[4];
    const float* bgate  = (const float*)d_in[5];
    const float* Worig  = (const float*)d_in[6];
    const float* borig  = (const float*)d_in[7];
    const float* W1     = (const float*)d_in[8];
    const float* b1     = (const float*)d_in[9];
    const float* W2     = (const float*)d_in[10];
    const float* b2     = (const float*)d_in[11];
    const float* gamma  = (const float*)d_in[12];
    float* out = (float*)d_out;

    const int N = in_sizes[0] / D_IN;
    const int E = in_sizes[1] / 2;

    // workspace carve-up (all offsets 256B aligned)
    char* ws = (char*)d_ws;
    size_t off = 0;
    float*  Hsum  = (float*) (ws + off); off += (size_t)N * C4 * sizeof(float);
    __bf16* Hall  = (__bf16*)(ws + off); off += (size_t)N * C4 * sizeof(__bf16);
    __bf16* xb    = (__bf16*)(ws + off); off += (size_t)N * D_IN * sizeof(__bf16);
    __bf16* zb    = xb;  // alias: xb dead after gemm1, zb written by epilogue
    __bf16* WcatT = (__bf16*)(ws + off); off += (size_t)C4 * D_IN * sizeof(__bf16);
    float*  bcat  = (float*) (ws + off); off += 2048;
    __bf16* W1T   = (__bf16*)(ws + off); off += (size_t)H_DIM * 256 * sizeof(__bf16);
    (void)ws_size; (void)n_in; (void)out_size;

    // 1. zero fp32 accumulator
    long n4 = (long)N * C4 / 4;
    k_zero_f4<<<16384, 256, 0, stream>>>((float4*)Hsum, n4);
    // 2. pack weights
    k_pack_wcat<<<(C4 * D_IN + 255) / 256, 256, 0, stream>>>(Word, bord, Worig, borig, WcatT, bcat);
    k_pack_w1<<<(H_DIM * 256 + 255) / 256, 256, 0, stream>>>(W1, W1T);
    // 3. x -> bf16
    k_cvt_x<<<16384, 256, 0, stream>>>(x, xb, (long)N * D_IN);
    // 4. fused 4-way GEMM (bf16 output)
    int mtiles = (N + 15) / 16;
    k_gemm1<<<mtiles, 256, 0, stream>>>(xb, WcatT, bcat, Hall, N);
    // 5. SpMM (bf16 gather, fp32 atomic scatter-add), one wave per edge
    long spmm_threads = (long)E * 32;
    k_spmm<<<(unsigned)((spmm_threads + 255) / 256), 256, 0, stream>>>(eidx, Hall, Hsum, E);
    // 6. gating epilogue, one wave per node
    long epi_threads = (long)N * 32;
    k_epilogue<<<(unsigned)((epi_threads + 255) / 256), 256, 0, stream>>>(Hsum, Wgate, bgate, gamma, zb, N);
    // 7. classifier GEMM + 128x2 projection
    k_gemm2<<<(mtiles + 3) / 4, 128, 0, stream>>>(zb, W1T, b1, W2, b2, out, N);
}
